// ToMeAttention_13769665151606
// MI455X (gfx1250) — compile-verified, matrix operand order checked by reading, not run on hardware
//
#include <hip/hip_runtime.h>

// ---------------------------------------------------------------------------
// ToMe attention for MI455X (gfx1250): bf16 WMMA + TDM (tensor_load_to_lds)
//   B=8, N=2048, C=768, H=12, HD=64, SCALE=1/8
// Pipeline: convert -> gemm(q) -> gemm(k) -> gemm(v) -> flash attn -> gemm(o)
//           -> kmean
// ---------------------------------------------------------------------------

#define B_  8
#define N_  2048
#define C_  768
#define H_  12
#define HD_ 64
#define SCALE_ 0.125f

typedef __bf16 bf16;
typedef __attribute__((ext_vector_type(16))) __bf16 v16bf;
typedef __attribute__((ext_vector_type(8)))  float  v8f;
typedef __attribute__((ext_vector_type(4)))  unsigned int v4u;
typedef __attribute__((ext_vector_type(4)))  unsigned int tdm_g0_t;
typedef __attribute__((ext_vector_type(8)))  int          tdm_g1_t;
typedef __attribute__((ext_vector_type(4)))  int          tdm_g2_t;
typedef __attribute__((ext_vector_type(8)))  int          tdm_g3x_t;

#if __has_builtin(__builtin_amdgcn_tensor_load_to_lds) && \
    __has_builtin(__builtin_amdgcn_s_wait_tensorcnt)
#define USE_TDM 1
#else
#define USE_TDM 0
#endif

__device__ __forceinline__ bf16 f2bf(float f) {
  unsigned u = __builtin_bit_cast(unsigned, f);
  unsigned short h = (unsigned short)((u + 0x7FFFu + ((u >> 16) & 1u)) >> 16);
  return __builtin_bit_cast(bf16, h);
}
__device__ __forceinline__ float bf2f(bf16 h) {
  unsigned u = ((unsigned)__builtin_bit_cast(unsigned short, h)) << 16;
  return __builtin_bit_cast(float, u);
}
__device__ __forceinline__ v8f wmma_bf16(v16bf a, v16bf b, v8f c) {
  return __builtin_amdgcn_wmma_f32_16x16x32_bf16(false, a, false, b,
                                                 (short)0, c, false, false);
}
// Load one 16x32 bf16 fragment row from a 16B-aligned base:
// lane 0-15 -> K {0..7, 16..23}; lane 16-31 -> K {8..15, 24..31} (hsel=8).
__device__ __forceinline__ v16bf load_frag(const bf16* p) {
  v16bf f;
  ((v4u*)&f)[0] = *(const v4u*)p;
  ((v4u*)&f)[1] = *(const v4u*)(p + 16);
  return f;
}

#if USE_TDM
// TDM: DMA a 2-D tile (tile_d1 rows x tile_d0 elems of 2-byte data) from a
// row-major global tensor (row stride = stride0 elems) into contiguous LDS.
// D# layout per CDNA5 ISA 8.3/8.4.  Wave-uniform; EXEC ignored.
// This toolchain's builtin takes 6 args:
//   (uint32x4 g0, int32x8 g1, int32x4 g2, int32x4 g3, int32x8, i32 cpol)
__device__ __forceinline__ void tdm_load_2d_bf16(
    unsigned lds_addr, const void* gaddr, unsigned tensor_d0,
    unsigned tensor_d1, unsigned tile_d0, unsigned tile_d1, unsigned stride0) {
  unsigned long long ga = (unsigned long long)(size_t)gaddr;
  tdm_g0_t g0;
  g0[0] = 1u;                                   // count=1, user mode
  g0[1] = lds_addr;                             // [63:32] lds byte addr
  g0[2] = (unsigned)(ga & 0xFFFFFFFFu);         // [95:64] global addr lo
  g0[3] = (unsigned)((ga >> 32) & 0x01FFFFFFu)  // [120:96] global addr hi
          | (2u << 30);                         // [127:126] type=2 (image)
  tdm_g1_t g1;
  g1[0] = (int)(1u << 16);                      // mask=0, data_size=1 (2B)
  g1[1] = (int)((tensor_d0 & 0xFFFFu) << 16);   // [63:48] tensor_dim0 lo16
  g1[2] = (int)(((tensor_d0 >> 16) & 0xFFFFu) | ((tensor_d1 & 0xFFFFu) << 16));
  g1[3] = (int)(((tensor_d1 >> 16) & 0xFFFFu) | ((tile_d0 & 0xFFFFu) << 16));
  g1[4] = (int)(tile_d1 & 0xFFFFu);             // tile_dim1; tile_dim2=0
  g1[5] = (int)stride0;                         // tensor_dim0_stride lo32
  g1[6] = 0;                                    // stride0 hi16, dim1_stride=0
  g1[7] = 0;
  tdm_g2_t gz = {0, 0, 0, 0};                   // 2-D: groups 2/3 disabled
  tdm_g3x_t gz8 = {0, 0, 0, 0, 0, 0, 0, 0};
  __builtin_amdgcn_tensor_load_to_lds(g0, g1, gz, gz, gz8, 0);
}
__device__ __forceinline__ unsigned lds_off(const void* p) {
  return (unsigned)(size_t)p;  // low 32 bits of generic ptr = LDS offset
}
#endif

// ---------------------------------------------------------------------------
// Kernel 0: x -> bf16, logsize = log(size)
// ---------------------------------------------------------------------------
__global__ __launch_bounds__(256) void convert_kernel(
    const float* __restrict__ x, const float* __restrict__ size,
    bf16* __restrict__ x_bf, float* __restrict__ logsize, int total) {
  int i = blockIdx.x * 256 + threadIdx.x;
  if (i < total) x_bf[i] = f2bf(x[i]);
  if (i < B_ * N_) logsize[i] = __logf(size[i]);
}

// ---------------------------------------------------------------------------
// Kernel 1: GEMM  D = A(bf16)[M,768] * W(f32)[768out,768in]^T + bias
// Block: 256 thr = 8 waves; tile 128 rows x 64 cols; K step 32.
// A tile staged by TDM (wave 0); W tile converted f32->bf16 by all waves.
// MODE 0: scatter bf16 into [B,H,N,64]   MODE 1: f32 row-major [M,768]
// ---------------------------------------------------------------------------
template <int MODE>
__global__ __launch_bounds__(256) void gemm_bf16_kernel(
    const bf16* __restrict__ A, const float* __restrict__ W,
    const float* __restrict__ bias, void* __restrict__ D) {
  __shared__ bf16 As[128 * 32];  // 8 KB
  __shared__ bf16 Bs[64 * 32];   // 4 KB

  const int lane = threadIdx.x & 31, wave = threadIdx.x >> 5;
  const int lrow = lane & 15;
  const int hsel = (lane < 16) ? 0 : 8;
  const int row_base = blockIdx.x * 128;
  const int col_base = blockIdx.y * 64;

  v8f acc[4];
#pragma unroll
  for (int t = 0; t < 4; ++t)
#pragma unroll
    for (int r = 0; r < 8; ++r) acc[t][r] = 0.f;

  for (int kk = 0; kk < C_; kk += 32) {
    // stage A tile 128x32 bf16
#if USE_TDM
    if (wave == 0)
      tdm_load_2d_bf16(lds_off(As), A + (size_t)row_base * C_ + kk,
                       /*tensor_d0=*/C_, /*tensor_d1=*/B_ * N_,
                       /*tile_d0=*/32, /*tile_d1=*/128, /*stride0=*/C_);
#else
#pragma unroll
    for (int c = threadIdx.x; c < 512; c += 256) {
      int row = c >> 2, off = (c & 3) * 8;
      *(v4u*)(As + row * 32 + off) =
          *(const v4u*)(A + (size_t)(row_base + row) * C_ + kk + off);
    }
#endif
    // stage W tile 64x32: f32 -> bf16 (8 elems per thread)
    {
      int c = threadIdx.x;  // 256 chunks exactly
      int row = c >> 2, off = (c & 3) * 8;
      const float* src = W + (size_t)(col_base + row) * C_ + kk + off;
      __align__(16) bf16 tmp[8];
#pragma unroll
      for (int i = 0; i < 8; ++i) tmp[i] = f2bf(src[i]);
      *(v4u*)(Bs + row * 32 + off) = *(v4u*)tmp;
    }
    if (kk + 32 < C_)
      __builtin_prefetch(W + (size_t)(col_base + (threadIdx.x >> 2)) * C_ + kk + 32, 0, 0);
#if USE_TDM
    if (wave == 0) __builtin_amdgcn_s_wait_tensorcnt(0);
#endif
    __syncthreads();

    v16bf a = load_frag(As + (wave * 16 + lrow) * 32 + hsel);
#pragma unroll
    for (int t = 0; t < 4; ++t) {
      v16bf b = load_frag(Bs + (t * 16 + lrow) * 32 + hsel);
      acc[t] = wmma_bf16(a, b, acc[t]);
    }
    __syncthreads();
  }

  // epilogue: C/D layout -> VGPR r holds (row r | row r+8, col lane%16)
#pragma unroll
  for (int t = 0; t < 4; ++t) {
    int colg = col_base + t * 16 + lrow;
    float bv = bias[colg];
#pragma unroll
    for (int r = 0; r < 8; ++r) {
      int m = r + hsel;
      int rowg = row_base + wave * 16 + m;
      float val = acc[t][r] + bv;
      if (MODE == 0) {
        int b = rowg / N_, n = rowg % N_;
        int h = colg / HD_, hd = colg % HD_;
        ((bf16*)D)[(((size_t)b * H_ + h) * N_ + n) * HD_ + hd] = f2bf(val);
      } else {
        ((float*)D)[(size_t)rowg * C_ + colg] = val;
      }
    }
  }
}

// ---------------------------------------------------------------------------
// Kernel 2: flash attention, one (b,h) x 128 queries per block.
// softmax(QK^T * SCALE + log(size)[key]) @ V  -> bf16 [B,N,C]
// K tile staged via TDM; V tile transposed manually (TDM cannot transpose).
// ---------------------------------------------------------------------------
__global__ __launch_bounds__(256) void attn_kernel(
    const bf16* __restrict__ Q, const bf16* __restrict__ K,
    const bf16* __restrict__ V, const float* __restrict__ logsize,
    bf16* __restrict__ O) {
  __shared__ bf16 kt[64 * 64];      // [key][d]    8 KB
  __shared__ bf16 vt[64 * 64];      // [d][key]    8 KB (transposed)
  __shared__ bf16 ps[8 * 16 * 64];  // per-wave P 16 KB

  const int bh = blockIdx.y;
  const int b = bh / H_, h = bh % H_;
  const int lane = threadIdx.x & 31, wave = threadIdx.x >> 5;
  const int lrow = lane & 15;
  const int hsel = (lane < 16) ? 0 : 8;
  const int q0 = blockIdx.x * 128 + wave * 16;

  // Q fragments (16 queries x 64 d), loaded once
  const bf16* qptr = Q + ((size_t)bh * N_ + q0) * HD_;
  v16bf aq0 = load_frag(qptr + lrow * HD_ + 0 + hsel);
  v16bf aq1 = load_frag(qptr + lrow * HD_ + 32 + hsel);

  v8f acc[4];
  float mrow[8], lsum[8];
#pragma unroll
  for (int j = 0; j < 4; ++j)
#pragma unroll
    for (int r = 0; r < 8; ++r) acc[j][r] = 0.f;
#pragma unroll
  for (int r = 0; r < 8; ++r) { mrow[r] = -1e30f; lsum[r] = 0.f; }

  const bf16* kbase = K + (size_t)bh * N_ * HD_;
  const bf16* vbase = V + (size_t)bh * N_ * HD_;
  const float* ls = logsize + (size_t)b * N_;
  bf16* pw = ps + wave * (16 * 64);

  for (int kb = 0; kb < N_; kb += 64) {
    // stage K tile row-major (TDM DMA, issued by wave 0)
#if USE_TDM
    if (wave == 0)
      tdm_load_2d_bf16(lds_off(kt), kbase + (size_t)kb * HD_,
                       /*tensor_d0=*/HD_, /*tensor_d1=*/N_,
                       /*tile_d0=*/HD_, /*tile_d1=*/64, /*stride0=*/HD_);
#else
#pragma unroll
    for (int c = threadIdx.x; c < 512; c += 256) {
      int key = c >> 3, off = (c & 7) * 8;
      *(v4u*)(kt + key * 64 + off) =
          *(const v4u*)(kbase + (size_t)(kb + key) * HD_ + off);
    }
#endif
    // stage V tile transposed: vt[d][key]
#pragma unroll
    for (int c = threadIdx.x; c < 512; c += 256) {
      int key = c >> 3, off = (c & 7) * 8;
      __align__(16) bf16 tmp[8];
      *(v4u*)tmp = *(const v4u*)(vbase + (size_t)(kb + key) * HD_ + off);
#pragma unroll
      for (int i = 0; i < 8; ++i) vt[(off + i) * 64 + key] = tmp[i];
    }
#if USE_TDM
    if (wave == 0) __builtin_amdgcn_s_wait_tensorcnt(0);
#endif
    __syncthreads();

    // scores: 4 tiles of 16 queries x 16 keys
    v8f s[4];
#pragma unroll
    for (int t = 0; t < 4; ++t) {
      const bf16* p = kt + (t * 16 + lrow) * 64 + hsel;
      v16bf bk0 = load_frag(p);
      v16bf bk1 = load_frag(p + 32);
      v8f z;
#pragma unroll
      for (int r = 0; r < 8; ++r) z[r] = 0.f;
      z = wmma_bf16(aq0, bk0, z);
      z = wmma_bf16(aq1, bk1, z);
      float lsv = ls[kb + t * 16 + lrow];
#pragma unroll
      for (int r = 0; r < 8; ++r) s[t][r] = z[r] * SCALE_ + lsv;
    }

    // online softmax per output row (VGPR r => rows r / r+8 per lane half)
#pragma unroll
    for (int r = 0; r < 8; ++r) {
      float mx = fmaxf(fmaxf(s[0][r], s[1][r]), fmaxf(s[2][r], s[3][r]));
#pragma unroll
      for (int d = 1; d < 16; d <<= 1) mx = fmaxf(mx, __shfl_xor(mx, d, 32));
      float mnew = fmaxf(mrow[r], mx);
      float alpha = __expf(mrow[r] - mnew);
      mrow[r] = mnew;
      float ssum = 0.f;
#pragma unroll
      for (int t = 0; t < 4; ++t) {
        float pv = __expf(s[t][r] - mnew);
        s[t][r] = pv;
        ssum += pv;
      }
#pragma unroll
      for (int d = 1; d < 16; d <<= 1) ssum += __shfl_xor(ssum, d, 32);
      lsum[r] = lsum[r] * alpha + ssum;
#pragma unroll
      for (int j = 0; j < 4; ++j) acc[j][r] *= alpha;
    }

    // P: C/D layout -> A layout via per-wave LDS round-trip
#pragma unroll
    for (int t = 0; t < 4; ++t)
#pragma unroll
      for (int r = 0; r < 8; ++r)
        pw[(r + hsel) * 64 + t * 16 + lrow] = f2bf(s[t][r]);

    v16bf ap0 = load_frag(pw + lrow * 64 + 0 + hsel);
    v16bf ap1 = load_frag(pw + lrow * 64 + 32 + hsel);

    // acc += P @ V
#pragma unroll
    for (int j = 0; j < 4; ++j) {
      const bf16* p = vt + (j * 16 + lrow) * 64 + hsel;
      v16bf bv0 = load_frag(p);
      v16bf bv1 = load_frag(p + 32);
      acc[j] = wmma_bf16(ap0, bv0, acc[j]);
      acc[j] = wmma_bf16(ap1, bv1, acc[j]);
    }
    __syncthreads();
  }

  // epilogue: normalize, scatter into [B,N,C] bf16
#pragma unroll
  for (int j = 0; j < 4; ++j)
#pragma unroll
    for (int r = 0; r < 8; ++r) {
      int row = q0 + r + hsel;
      int col = h * HD_ + j * 16 + lrow;
      O[((size_t)b * N_ + row) * C_ + col] = f2bf(acc[j][r] / lsum[r]);
    }
}

// ---------------------------------------------------------------------------
// Kernel 3: k.mean(axis=heads) -> f32 [B,N,64]
// ---------------------------------------------------------------------------
__global__ __launch_bounds__(256) void kmean_kernel(const bf16* __restrict__ Kb,
                                                    float* __restrict__ out) {
  int i = blockIdx.x * 256 + threadIdx.x;
  if (i >= B_ * N_ * HD_) return;
  int d = i % HD_;
  int n = (i / HD_) % N_;
  int b = i / (HD_ * N_);
  float s = 0.f;
#pragma unroll
  for (int h = 0; h < H_; ++h)
    s += bf2f(Kb[(((size_t)b * H_ + h) * N_ + n) * HD_ + d]);
  out[i] = s * (1.f / H_);
}

// ---------------------------------------------------------------------------
extern "C" void kernel_launch(void* const* d_in, const int* in_sizes, int n_in,
                              void* d_out, int out_size, void* d_ws,
                              size_t ws_size, hipStream_t stream) {
  const float* x    = (const float*)d_in[0];
  const float* size = (const float*)d_in[1];
  const float* q_w  = (const float*)d_in[2];
  const float* q_b  = (const float*)d_in[3];
  const float* k_w  = (const float*)d_in[4];
  const float* k_b  = (const float*)d_in[5];
  const float* v_w  = (const float*)d_in[6];
  const float* v_b  = (const float*)d_in[7];
  const float* o_w  = (const float*)d_in[8];
  const float* o_b  = (const float*)d_in[9];

  const size_t nx   = (size_t)B_ * N_ * C_;        // 12.58M elems
  const size_t nqkv = (size_t)B_ * H_ * N_ * HD_;  // == nx

  bf16* x_bf = (bf16*)d_ws;
  bf16* qb   = x_bf + nx;
  bf16* kb   = qb + nqkv;
  bf16* vb   = kb + nqkv;
  bf16* ob   = vb + nqkv;               // pre-o-proj attention output (bf16)
  float* logsize = (float*)(ob + nx);   // B*N floats

  convert_kernel<<<(unsigned)((nx + 255) / 256), 256, 0, stream>>>(
      x, size, x_bf, logsize, (int)nx);

  dim3 gg(128, 12);  // 16384/128 row tiles x 768/64 col tiles
  gemm_bf16_kernel<0><<<gg, 256, 0, stream>>>(x_bf, q_w, q_b, qb);
  gemm_bf16_kernel<0><<<gg, 256, 0, stream>>>(x_bf, k_w, k_b, kb);
  gemm_bf16_kernel<0><<<gg, 256, 0, stream>>>(x_bf, v_w, v_b, vb);

  dim3 ga(N_ / 128, B_ * H_);  // 16 x 96
  attn_kernel<<<ga, 256, 0, stream>>>(qb, kb, vb, logsize, ob);

  gemm_bf16_kernel<1><<<gg, 256, 0, stream>>>(ob, o_w, o_b, d_out);

  kmean_kernel<<<(B_ * N_ * HD_ + 255) / 256, 256, 0, stream>>>(
      kb, (float*)d_out + nx);
}
